// ENNEncoder_53128745452308
// MI455X (gfx1250) — matrix operand order repaired and checked by reading, not compiled, not run on hardware
//
#include <hip/hip_runtime.h>

typedef float     v2f  __attribute__((ext_vector_type(2)));
typedef float     v8f  __attribute__((ext_vector_type(8)));
typedef _Float16  v2h  __attribute__((ext_vector_type(2)));
typedef _Float16  v16h __attribute__((ext_vector_type(16)));

// ---------------- problem sizes ----------------
static constexpr int B_  = 16;
static constexpr int C_  = 129;
static constexpr int T_  = 2000;
static constexpr int NF_ = 40;
static constexpr int ED_ = 64;
static constexpr int K_  = 16;
static constexpr int TP_ = 2048;              // time padded to 16*128

// stage-1 WMMA geometry: per channel c, tconv is one K=32 chunk (25 taps padded),
// sconv contraction over f is two K=32 chunks (40 rows padded to 64), M padded to 48.
static constexpr int APK_TILES = C_ * 2 * 3;            // (c, chunk, m-tile)
static constexpr int APK_ELEMS = APK_TILES * 32 * 16;   // halves: per tile 32 lanes x 16
static constexpr int APKT_ELEMS = 3 * 32 * 16;          // tconv A tiles

// ---------------- workspace layout (floats) ----------------
static constexpr size_t OFF_APK  = 0;                                  // f16 sconv A pack
static constexpr size_t SZ_APK   = APK_ELEMS / 2;                      // 198144 floats
static constexpr size_t OFF_APKT = OFF_APK + SZ_APK;                   // f16 tconv A pack
static constexpr size_t SZ_APKT  = APKT_ELEMS / 2;                     // 768
static constexpr size_t OFF_TB2  = OFF_APKT + SZ_APKT;                 // 40
static constexpr size_t OFF_SA   = OFF_TB2 + NF_;                      // 40
static constexpr size_t OFF_SB2  = OFF_SA  + NF_;                      // 40
static constexpr size_t OFF_PA   = OFF_SB2 + NF_;                      // 80
static constexpr size_t OFF_PB2  = OFF_PA  + 2 * NF_;                  // 80
static constexpr size_t OFF_S    = ((OFF_PB2 + 2 * NF_) + 7) & ~(size_t)7;
static constexpr size_t SZ_S     = (size_t)B_ * NF_ * TP_;             // 1.31M
static constexpr size_t OFF_FEAT = OFF_S + SZ_S;                       // 512000
// total ~2.02M floats = ~8.1 MB of workspace

static constexpr int XS_W      = 160;   // 128 outputs + 12 left halo + 20 right (K padded to 32)
static constexpr int XS_STRIDE = 161;
static constexpr int ELS_STRIDE = 19;   // v2h units per staging row (bank-conflict padding)
static constexpr int ELS_WSZ    = 32 * ELS_STRIDE;   // rows q=0..31 per wave (q>=20 stay zero)

// K index inside a 16x32 f16 A-operand for (half-wave hi, vector slot kidx):
// hi=0 -> K = {0..15}[kidx] ; hi=1 -> K = {16..31}[kidx]
__device__ __host__ inline int a_pattern_K(int hi, int kidx) {
    return kidx + hi * 16;
}

// ================= prep: fold BN, build f16 A-operand packs =================
__global__ void prep_small(const float* __restrict__ tconv_w, const float* __restrict__ tconv_b,
                           const float* __restrict__ tbn_g, const float* __restrict__ tbn_b,
                           const float* __restrict__ tbn_m, const float* __restrict__ tbn_v,
                           const float* __restrict__ sconv_b,
                           const float* __restrict__ sbn_g, const float* __restrict__ sbn_b,
                           const float* __restrict__ sbn_m, const float* __restrict__ sbn_v,
                           const float* __restrict__ pw_b,
                           const float* __restrict__ pbn_g, const float* __restrict__ pbn_b,
                           const float* __restrict__ pbn_m, const float* __restrict__ pbn_v,
                           _Float16* __restrict__ apkT, float* __restrict__ tb2,
                           float* __restrict__ sa,  float* __restrict__ sb2,
                           float* __restrict__ pa,  float* __restrict__ pb2)
{
    const int tid = threadIdx.x;
    // tconv A tiles in f16 register layout: A[M=f][K=tap], BN scale folded
    for (int i = tid; i < APKT_ELEMS; i += blockDim.x) {
        int kidx = i & 15;
        int laneFlat = (i >> 4) & 31;
        int m = i >> 9;
        int hi = laneFlat >> 4, nn = laneFlat & 15;
        int Kt = a_pattern_K(hi, kidx);
        int f = m * 16 + nn;
        float v = 0.f;
        if (f < NF_ && Kt < 25) {
            float inv = tbn_g[f] * rsqrtf(tbn_v[f] + 1e-5f);
            v = tconv_w[f * 25 + Kt] * inv;
        }
        apkT[i] = (_Float16)v;
    }
    if (tid < NF_) {
        float tinv = tbn_g[tid] * rsqrtf(tbn_v[tid] + 1e-5f);
        tb2[tid] = (tconv_b[tid] - tbn_m[tid]) * tinv + tbn_b[tid];
        float sinv = sbn_g[tid] * rsqrtf(sbn_v[tid] + 1e-5f);
        sa[tid]  = sinv;
        sb2[tid] = (sconv_b[tid] - sbn_m[tid]) * sinv + sbn_b[tid];
    }
    if (tid < 2 * NF_) {
        float pinv = pbn_g[tid] * rsqrtf(pbn_v[tid] + 1e-5f);
        pa[tid]  = pinv;
        pb2[tid] = (pw_b[tid] - pbn_m[tid]) * pinv + pbn_b[tid];
    }
}

// sconv_w(g,f,c) -> f16 A-operand pack, per-lane contiguous 16 halves (2x b128 per lane)
__global__ void prep_apk(const float* __restrict__ sconv_w, _Float16* __restrict__ apk)
{
    int idx = blockIdx.x * blockDim.x + threadIdx.x;
    if (idx >= APK_ELEMS) return;
    int kidx = idx & 15;
    int laneFlat = (idx >> 4) & 31;
    int tile = idx >> 9;
    int m  = tile % 3;
    int t2 = tile / 3;
    int ch = t2 & 1;
    int c  = t2 >> 1;
    int hi = laneFlat >> 4, nn = laneFlat & 15;
    int Kt = a_pattern_K(hi, kidx);
    int f = ch * 32 + Kt;          // reduction row = input feature
    int g = m * 16 + nn;           // output feature
    float v = (f < NF_ && g < NF_) ? sconv_w[((size_t)g * NF_ + f) * C_ + c] : 0.f;
    apk[idx] = (_Float16)v;
}

// ================= stage 1: tconv(WMMA f16) + BN + ELU + sconv(WMMA f16) + BN + ELU =================
// grid: (16 t-tiles of 128, B). block: 256 = 8 waves, each owns 16 time columns.
__global__ __launch_bounds__(256) void stage1(const float* __restrict__ x,
                                              const _Float16* __restrict__ apk,
                                              const _Float16* __restrict__ apkT,
                                              const float* __restrict__ tb2,
                                              const float* __restrict__ sa,
                                              const float* __restrict__ sb2,
                                              float* __restrict__ sbuf)
{
    extern __shared__ float smem[];
    float* xs  = smem;                                   // [129][161] f32 x tile (+zero halo)
    v2h*   els = (v2h*)(smem + (size_t)C_ * XS_STRIDE);  // per-wave f16-pair staging, 8*ELS_WSZ

    const int tid = threadIdx.x;
    const int b   = blockIdx.y;
    const int t0  = blockIdx.x * 128;

    for (int i = tid; i < C_ * XS_W; i += 256) {
        int c = i / XS_W, j = i - c * XS_W;
        int t = t0 - 12 + j;
        xs[c * XS_STRIDE + j] = (t >= 0 && t < T_) ? x[((size_t)b * C_ + c) * T_ + t] : 0.f;
    }
    {
        v2h z = {};
        for (int i = tid; i < 8 * ELS_WSZ; i += 256) els[i] = z;  // rows q>=20 must read as 0
    }
    __syncthreads();

    const int lane = tid & 31;
    const int wv   = tid >> 5;
    const int n    = lane & 15;
    const int hi   = lane >> 4;
    const int tloc = wv * 16 + n;
    const int wbase = wv * ELS_WSZ;

    // persistent tconv A tiles (f16 A layout, BN-folded)
    const v16h* apkTv = (const v16h*)apkT;
    const v16h aT0 = apkTv[0 * 32 + lane];
    const v16h aT1 = apkTv[1 * 32 + lane];
    const v16h aT2 = apkTv[2 * 32 + lane];

    // tconv C operand = folded bias broadcast along t (D rows f = m*16 + r + 8*hi)
    v8f c0, c1, c2;
    #pragma unroll
    for (int r = 0; r < 8; ++r) {
        c0[r] = tb2[r + 8 * hi];
        c1[r] = tb2[16 + r + 8 * hi];
        c2[r] = hi ? 0.f : tb2[32 + r];
    }

    v8f acc0 = {}; v8f acc1 = {}; v8f acc2 = {};
    const v16h* apkv = (const v16h*)apk;

    for (int c = 0; c < C_; ++c) {
        // ---- tconv B tile: B[K=tap][N=t], f16 B layout (lanes hi get K=16..31) ----
        const float* xrow = xs + c * XS_STRIDE + tloc + hi * 16;
        v16h bt;
        #pragma unroll
        for (int j = 0; j < 8; ++j) {
            bt[2 * j]     = (_Float16)xrow[2 * j];
            bt[2 * j + 1] = (_Float16)xrow[2 * j + 1];
        }
        v8f d0 = __builtin_amdgcn_wmma_f32_16x16x32_f16(false, aT0, false, bt, (short)0, c0, false, false);
        v8f d1 = __builtin_amdgcn_wmma_f32_16x16x32_f16(false, aT1, false, bt, (short)0, c1, false, false);
        v8f d2 = __builtin_amdgcn_wmma_f32_16x16x32_f16(false, aT2, false, bt, (short)0, c2, false, false);

        // ---- ELU, pack consecutive-f pairs to f16, stage into B-layout rows ----
        #pragma unroll
        for (int m = 0; m < 3; ++m) {
            const v8f& dm = (m == 0) ? d0 : ((m == 1) ? d1 : d2);
            #pragma unroll
            for (int rp = 0; rp < 4; ++rp) {
                float e0 = dm[2 * rp], e1 = dm[2 * rp + 1];
                e0 = e0 > 0.f ? e0 : (__expf(e0) - 1.f);
                e1 = e1 > 0.f ? e1 : (__expf(e1) - 1.f);
                int q = m * 8 + 4 * hi + rp;                 // f-pair index = f/2
                if (q < 20) {                                // f < 40 only
                    v2h pr; pr[0] = (_Float16)e0; pr[1] = (_Float16)e1;
                    els[wbase + q * ELS_STRIDE + n] = pr;
                }
            }
        }
        asm volatile("s_wait_dscnt 0" ::: "memory");         // wave-local store->load ordering

        // ---- sconv: two K=32 chunks over f, three M tiles of g ----
        const size_t tbase = (size_t)c * 6;                  // (c*2+ch)*3 tiles
        #pragma unroll
        for (int ch = 0; ch < 2; ++ch) {
            v16h bs;
            #pragma unroll
            for (int j = 0; j < 8; ++j) {
                v2h pr = els[wbase + (ch * 16 + 8 * hi + j) * ELS_STRIDE + n];
                bs[2 * j] = pr[0]; bs[2 * j + 1] = pr[1];
            }
            const v16h* ap = apkv + (tbase + (size_t)ch * 3) * 32 + lane;
            __builtin_prefetch(ap + 6 * 32, 0, 1);           // next c's A panel
            acc0 = __builtin_amdgcn_wmma_f32_16x16x32_f16(false, ap[0],      false, bs, (short)0, acc0, false, false);
            acc1 = __builtin_amdgcn_wmma_f32_16x16x32_f16(false, ap[32],     false, bs, (short)0, acc1, false, false);
            acc2 = __builtin_amdgcn_wmma_f32_16x16x32_f16(false, ap[64],     false, bs, (short)0, acc2, false, false);
        }
    }

    // epilogue: sconv BN + ELU, write s[b][g][t] (t in [2000,2048) forced to 0)
    const int t  = t0 + tloc;
    const bool tv = (t < T_);
    #pragma unroll
    for (int m = 0; m < 3; ++m) {
        const v8f& a = (m == 0) ? acc0 : ((m == 1) ? acc1 : acc2);
        #pragma unroll
        for (int r = 0; r < 8; ++r) {
            int g = m * 16 + r + 8 * hi;
            if (g < NF_) {
                float v = a[r] * sa[g] + sb2[g];
                v = v > 0.f ? v : (__expf(v) - 1.f);
                sbuf[((size_t)b * NF_ + g) * TP_ + t] = tv ? v : 0.f;
            }
        }
    }
}

// ================= stage 2: dwconv + pwconv + BN + ELU + avgpool4 + projection =================
__global__ __launch_bounds__(256) void stage2(const float* __restrict__ sbuf,
                                              const float* __restrict__ dw_w,
                                              const float* __restrict__ dw_b,
                                              const float* __restrict__ pw_w,
                                              const float* __restrict__ pa,
                                              const float* __restrict__ pb2,
                                              const float* __restrict__ proj_w,
                                              const float* __restrict__ proj_b,
                                              float* __restrict__ feats)
{
    extern __shared__ float smem[];
    float* st   = smem;                  // [40][272] input tile with halo
    float* dd   = st   + NF_ * 272;      // [40][256] depthwise result
    float* Pp   = dd   + NF_ * 256;      // [80][64]  pooled post-ELU
    float* dwWs = Pp   + 80 * 64;        // 600
    float* dwBs = dwWs + NF_ * 15;       // 40
    float* pwWs = dwBs + NF_;            // 3200
    float* pjWs = pwWs + 2 * NF_ * NF_;  // 5120
    float* pjBs = pjWs + ED_ * 2 * NF_;  // 64
    float* pas  = pjBs + ED_;            // 80
    float* pb2s = pas  + 2 * NF_;        // 80

    const int tid = threadIdx.x;
    const int b   = blockIdx.y;
    const int t0  = blockIdx.x * 256;
    const int tp0 = blockIdx.x * 64;

    for (int i = tid; i < NF_ * 271; i += 256) {
        int f = i / 271, j = i - f * 271;
        int t = t0 - 7 + j;
        st[f * 272 + j] = (t >= 0 && t < TP_) ? sbuf[((size_t)b * NF_ + f) * TP_ + t] : 0.f;
    }
    for (int i = tid; i < NF_ * 15; i += 256) dwWs[i] = dw_w[i];
    if (tid < NF_) dwBs[tid] = dw_b[tid];
    for (int i = tid; i < 2 * NF_ * NF_; i += 256) pwWs[i] = pw_w[i];
    for (int i = tid; i < ED_ * 2 * NF_; i += 256) pjWs[i] = proj_w[i];
    if (tid < ED_) pjBs[tid] = proj_b[tid];
    if (tid < 2 * NF_) { pas[tid] = pa[tid]; pb2s[tid] = pb2[tid]; }
    __syncthreads();

    for (int i = tid; i < NF_ * 256; i += 256) {
        int f = i >> 8, tl = i & 255;
        float sum = dwBs[f];
        #pragma unroll
        for (int k = 0; k < 15; ++k) sum = fmaf(st[f * 272 + tl + k], dwWs[f * 15 + k], sum);
        dd[i] = sum;
    }
    __syncthreads();

    for (int i = tid; i < 2 * NF_ * 64; i += 256) {
        int g = i >> 6, tp = i & 63;
        float accp = 0.f;
        #pragma unroll
        for (int t4 = 0; t4 < 4; ++t4) {
            int tl = tp * 4 + t4;
            float v = 0.f;
            for (int f = 0; f < NF_; ++f) v = fmaf(dd[f * 256 + tl], pwWs[g * NF_ + f], v);
            v = v * pas[g] + pb2s[g];
            v = v > 0.f ? v : (__expf(v) - 1.f);
            accp += v;
        }
        Pp[i] = accp * 0.25f;
    }
    __syncthreads();

    for (int i = tid; i < 64 * ED_; i += 256) {
        int tp = i >> 6, e = i & 63;
        int tpg = tp0 + tp;
        if (tpg < 500) {
            float v = pjBs[e];
            for (int g = 0; g < 2 * NF_; ++g) v = fmaf(Pp[g * 64 + tp], pjWs[e * 2 * NF_ + g], v);
            feats[((size_t)b * 500 + tpg) * ED_ + e] = v;
        }
    }
}

// ================= stage 3: sequential 500-step 2-cell scan =================
// thread=(batch,k); 16 k-lanes of a batch are a half-wave under wave32 ->
// matvec broadcast + layernorm entirely via shfl width 16.
__global__ __launch_bounds__(256) void rnn_kernel(const float* __restrict__ feats,
                                                  const float* __restrict__ wx0, const float* __restrict__ L0,
                                                  const float* __restrict__ ll0, const float* __restrict__ b0,
                                                  const float* __restrict__ g0, const float* __restrict__ be0,
                                                  const float* __restrict__ wx1, const float* __restrict__ L1,
                                                  const float* __restrict__ ll1, const float* __restrict__ b1,
                                                  const float* __restrict__ g1, const float* __restrict__ be1,
                                                  float* __restrict__ out)
{
    __shared__ float Wx0s[K_ * ED_];
    __shared__ float Wx1s[K_ * K_];
    __shared__ float E0s[K_ * K_];
    __shared__ float E1s[K_ * K_];
    __shared__ float xts[B_ * ED_];

    const int tid = threadIdx.x;
    const int bq  = tid >> 4;
    const int k   = tid & 15;

    for (int i = tid; i < K_ * ED_; i += 256) Wx0s[i] = wx0[i];
    {   // E = L@L.T - exp(ll)*I
        int row = tid >> 4, col = tid & 15;
        float e0 = 0.f, e1 = 0.f;
        for (int m = 0; m < K_; ++m) {
            e0 = fmaf(L0[row * K_ + m], L0[col * K_ + m], e0);
            e1 = fmaf(L1[row * K_ + m], L1[col * K_ + m], e1);
        }
        if (row == col) { e0 -= __expf(ll0[0]); e1 -= __expf(ll1[0]); }
        E0s[tid] = e0;
        E1s[tid] = e1;
        Wx1s[tid] = wx1[tid];
    }
    const float bb0 = b0[k], gg0 = g0[k], bbe0 = be0[k];
    const float bb1 = b1[k], gg1 = g1[k], bbe1 = be1[k];
    float h0 = 0.f, h1 = 0.f;
    __syncthreads();

    for (int t = 0; t < 500; ++t) {
        for (int i = tid; i < B_ * ED_; i += 256) {
            int bi = i >> 6, e = i & 63;
            xts[i] = feats[((size_t)bi * 500 + t) * ED_ + e];
        }
        __syncthreads();

        // ---- cell 0 ----
        float pre = bb0;
        #pragma unroll 8
        for (int e = 0; e < ED_; ++e) pre = fmaf(xts[bq * ED_ + e], Wx0s[k * ED_ + e], pre);
        #pragma unroll
        for (int j = 0; j < K_; ++j) pre = fmaf(__shfl(h0, j, 16), E0s[k * K_ + j], pre);
        float s = pre;
        #pragma unroll
        for (int off = 1; off < 16; off <<= 1) s += __shfl_xor(s, off, 16);
        float mu = s * 0.0625f;
        float d  = pre - mu;
        float vs = d * d;
        #pragma unroll
        for (int off = 1; off < 16; off <<= 1) vs += __shfl_xor(vs, off, 16);
        float inv = rsqrtf(vs * 0.0625f + 1e-5f);
        h0 = tanhf(d * inv * gg0 + bbe0);

        // ---- cell 1 ----
        pre = bb1;
        #pragma unroll
        for (int j = 0; j < K_; ++j) pre = fmaf(__shfl(h0, j, 16), Wx1s[k * K_ + j], pre);
        #pragma unroll
        for (int j = 0; j < K_; ++j) pre = fmaf(__shfl(h1, j, 16), E1s[k * K_ + j], pre);
        s = pre;
        #pragma unroll
        for (int off = 1; off < 16; off <<= 1) s += __shfl_xor(s, off, 16);
        mu = s * 0.0625f;
        d  = pre - mu;
        vs = d * d;
        #pragma unroll
        for (int off = 1; off < 16; off <<= 1) vs += __shfl_xor(vs, off, 16);
        inv = rsqrtf(vs * 0.0625f + 1e-5f);
        h1 = tanhf(d * inv * gg1 + bbe1);
        __syncthreads();
    }

    out[bq * K_ + k] = h1;                 // Z (16x16)
    if (k == 0) out[B_ * K_ + bq] = 1.0f;  // alpha: softmax over singleton axis == 1
}

// ================= host side =================
extern "C" void kernel_launch(void* const* d_in, const int* in_sizes, int n_in,
                              void* d_out, int out_size, void* d_ws, size_t ws_size,
                              hipStream_t stream)
{
    (void)in_sizes; (void)n_in; (void)out_size; (void)ws_size;

    const float* x       = (const float*)d_in[0];
    const float* tconv_w = (const float*)d_in[1];
    const float* tconv_b = (const float*)d_in[2];
    const float* tbn_g   = (const float*)d_in[3];
    const float* tbn_b   = (const float*)d_in[4];
    const float* tbn_m   = (const float*)d_in[5];
    const float* tbn_v   = (const float*)d_in[6];
    const float* sconv_w = (const float*)d_in[7];
    const float* sconv_b = (const float*)d_in[8];
    const float* sbn_g   = (const float*)d_in[9];
    const float* sbn_b   = (const float*)d_in[10];
    const float* sbn_m   = (const float*)d_in[11];
    const float* sbn_v   = (const float*)d_in[12];
    const float* dw_w    = (const float*)d_in[13];
    const float* dw_b    = (const float*)d_in[14];
    const float* pw_w    = (const float*)d_in[15];
    const float* pw_b    = (const float*)d_in[16];
    const float* pbn_g   = (const float*)d_in[17];
    const float* pbn_b   = (const float*)d_in[18];
    const float* pbn_m   = (const float*)d_in[19];
    const float* pbn_v   = (const float*)d_in[20];
    const float* proj_w  = (const float*)d_in[21];
    const float* proj_b  = (const float*)d_in[22];
    const float* wx0     = (const float*)d_in[23];
    const float* L0      = (const float*)d_in[24];
    const float* ll0     = (const float*)d_in[25];
    const float* b0      = (const float*)d_in[26];
    const float* ln0_g   = (const float*)d_in[27];
    const float* ln0_b   = (const float*)d_in[28];
    const float* wx1     = (const float*)d_in[29];
    const float* L1      = (const float*)d_in[30];
    const float* ll1     = (const float*)d_in[31];
    const float* b1      = (const float*)d_in[32];
    const float* ln1_g   = (const float*)d_in[33];
    const float* ln1_b   = (const float*)d_in[34];
    // d_in[35..37] (wq_w, wq_b, log_temp) unused: softmax over a size-1 axis is 1.

    float*    ws    = (float*)d_ws;
    _Float16* apk   = (_Float16*)(ws + OFF_APK);
    _Float16* apkT  = (_Float16*)(ws + OFF_APKT);
    float*    tb2   = ws + OFF_TB2;
    float*    sa    = ws + OFF_SA;
    float*    sb2   = ws + OFF_SB2;
    float*    pa    = ws + OFF_PA;
    float*    pb2   = ws + OFF_PB2;
    float*    sbuf  = ws + OFF_S;
    float*    featp = ws + OFF_FEAT;

    prep_small<<<1, 256, 0, stream>>>(tconv_w, tconv_b, tbn_g, tbn_b, tbn_m, tbn_v,
                                      sconv_b, sbn_g, sbn_b, sbn_m, sbn_v,
                                      pw_b, pbn_g, pbn_b, pbn_m, pbn_v,
                                      apkT, tb2, sa, sb2, pa, pb2);

    prep_apk<<<(APK_ELEMS + 255) / 256, 256, 0, stream>>>(sconv_w, apk);

    size_t smem1 = (size_t)(C_ * XS_STRIDE) * sizeof(float) + (size_t)(8 * ELS_WSZ) * sizeof(v2h);
    stage1<<<dim3(16, B_), 256, smem1, stream>>>(x, apk, apkT, tb2, sa, sb2, sbuf);

    size_t smem2 = (size_t)(NF_ * 272 + NF_ * 256 + 80 * 64 + NF_ * 15 + NF_ +
                            2 * NF_ * NF_ + ED_ * 2 * NF_ + ED_ + 2 * NF_ + 2 * NF_) * sizeof(float);
    stage2<<<dim3(8, B_), 256, smem2, stream>>>(sbuf, dw_w, dw_b, pw_w, pa, pb2,
                                                proj_w, proj_b, featp);

    rnn_kernel<<<1, 256, 0, stream>>>(featp, wx0, L0, ll0, b0, ln0_g, ln0_b,
                                      wx1, L1, ll1, b1, ln1_g, ln1_b,
                                      (float*)d_out);
}
